// EnhancedDepthwiseSeparableFFN_87892210745473
// MI455X (gfx1250) — matrix operand order repaired and checked by reading, not compiled, not run on hardware
//
#include <hip/hip_runtime.h>
#include <math.h>

// ---------------------------------------------------------------------------
// MI455X (gfx1250) implementation. wave32, WMMA bf16 16x16x32 f32-accum for
// the two pointwise GEMMs (FLOP-dominant); fp32 everywhere else.
// LDS tiles are stored in WMMA-fragment-friendly order so fragment loads are
// wide ds_load_b128s instead of scalar ds_load_u16 gathers.
// ---------------------------------------------------------------------------

typedef __attribute__((ext_vector_type(16))) __bf16 v16bf;
typedef __attribute__((ext_vector_type(8)))  __bf16 v8bf;
typedef __attribute__((ext_vector_type(4)))  __bf16 v4bf;
typedef __attribute__((ext_vector_type(8)))  float  v8f;

constexpr int   Bz  = 8;      // batch
constexpr int   Dd  = 256;    // model dim
constexpr int   Cc  = 1024;   // expanded channels
constexpr int   HW  = 1024;   // 32*32 spatial
constexpr float EPS = 1e-5f;
constexpr float RED_N = 8192.0f;   // B*H*W  (BN reduction count)

// ---- workspace layout (units: floats) -------------------------------------
constexpr size_t Y_OFF  = 0;                                   // gelu(conv1) raw  (8,1024,1024)
constexpr size_t T_OFF  = (size_t)Bz * Cc * HW;                // gelu(dyndw) raw  (8,1024,1024)
constexpr size_t V_OFF  = T_OFF * 2;                           // pre-BN3 proj     (8,256,1024)
constexpr size_t SUM1   = V_OFF + (size_t)Bz * Dd * HW;        // BN1 sums   [C]
constexpr size_t SSQ1   = SUM1 + Cc;
constexpr size_t GAPRAW = SSQ1 + Cc;                           // [B,C]
constexpr size_t SCALE1 = GAPRAW + (size_t)Bz * Cc;
constexpr size_t SHIFT1 = SCALE1 + Cc;
constexpr size_t H1BUF  = SHIFT1 + Cc;                         // [B,128]
constexpr size_t KWBUF  = H1BUF + (size_t)Bz * 128;            // [B,16] (9 used)
constexpr size_t SUM2   = KWBUF + (size_t)Bz * 16;
constexpr size_t SSQ2   = SUM2 + Cc;
constexpr size_t GAP2   = SSQ2 + Cc;                           // [B,C]
constexpr size_t MAX2   = GAP2 + (size_t)Bz * Cc;
constexpr size_t MIN2   = MAX2 + (size_t)Bz * Cc;
constexpr size_t SCALE2 = MIN2 + (size_t)Bz * Cc;
constexpr size_t SHIFT2 = SCALE2 + Cc;
constexpr size_t M2O    = SHIFT2 + Cc;                         // [B,C]  s*scale2
constexpr size_t A2O    = M2O + (size_t)Bz * Cc;               // [B,C]  s*shift2
constexpr size_t CAB    = A2O + (size_t)Bz * Cc;               // [B,D]  sum_c pw*a2
constexpr size_t AVGM   = CAB + (size_t)Bz * Dd;               // [B,HW]
constexpr size_t MAXM   = AVGM + (size_t)Bz * HW;
constexpr size_t SPM    = MAXM + (size_t)Bz * HW;
constexpr size_t SUM3   = SPM + (size_t)Bz * HW;               // [D]
constexpr size_t SSQ3   = SUM3 + Dd;
constexpr size_t SCALE3 = SSQ3 + Dd;
constexpr size_t SHIFT3 = SCALE3 + Dd;
constexpr size_t WS_END = SHIFT3 + Dd;

__device__ __forceinline__ float gelu_exact(float x) {
    return 0.5f * x * (1.0f + erff(x * 0.70710678118654752440f));
}

__device__ __forceinline__ v16bf cat8(v8bf lo, v8bf hi) {
    return __builtin_shufflevector(lo, hi, 0, 1, 2, 3, 4, 5, 6, 7,
                                            8, 9, 10, 11, 12, 13, 14, 15);
}

__device__ __forceinline__ v4bf pack4(float a, float b, float c, float d) {
    v4bf p;
    p[0] = (__bf16)a; p[1] = (__bf16)b; p[2] = (__bf16)c; p[3] = (__bf16)d;
    return p;
}

__global__ __launch_bounds__(256) void k_zero(float* __restrict__ p, int n) {
    int i = blockIdx.x * 256 + threadIdx.x;
    if (i < n) p[i] = 0.0f;
}

// ---------------------------------------------------------------------------
// GEMM1: y[b,c,s] = gelu( w1[c,:] . x[b,s,:] + b1[c] )
// Block tile 128(M) x 64(N), K=256 in steps of 32. 8 waves, each 2x2 WMMAs.
// As[m][k] row = 64B (frag = 2 x 16B runs); Bst[n][k] row = 64B (frag = 2 x 16B runs)
// Epilogue: write y, shuffle-reduce rows, atomics -> BN1 stats + GAP.
// ---------------------------------------------------------------------------
__global__ __launch_bounds__(256)
void k_gemm1(const float* __restrict__ x, const float* __restrict__ w1,
             const float* __restrict__ b1, float* __restrict__ ws) {
    __shared__ __align__(64) __bf16 As[128][32];   // [m][k]
    __shared__ __align__(64) __bf16 Bst[64][32];   // [n][k]  (B transposed)
    const int tid = threadIdx.x;
    const int b   = blockIdx.z;
    const int m0  = blockIdx.x * 128;
    const int n0  = blockIdx.y * 64;
    const int wave = tid >> 5, lane = tid & 31;
    const int lm = lane & 15, lh = lane >> 4;
    const int wm = (wave & 3) * 32;      // wave M offset
    const int wn = (wave >> 2) * 32;     // wave N offset

    v8f acc00 = {}, acc01 = {}, acc10 = {}, acc11 = {};

    for (int k0 = 0; k0 < 256; k0 += 32) {
        // A: 128x32, float4 global loads along k, 8B LDS stores
        #pragma unroll
        for (int i = 0; i < 4; ++i) {
            int g = tid + i * 256;               // float4 group id (0..1023)
            int m = g >> 3, k = (g & 7) * 4;
            float4 f = *reinterpret_cast<const float4*>(&w1[(size_t)(m0 + m) * 256 + (k0 + k)]);
            *reinterpret_cast<v4bf*>(&As[m][k]) = pack4(f.x, f.y, f.z, f.w);
        }
        // B: 32(k)x64(n); x is [b,s,c] contiguous along k=c -> float4 along k
        #pragma unroll
        for (int i = 0; i < 2; ++i) {
            int g = tid + i * 256;               // 0..511
            int n = g >> 3, k = (g & 7) * 4;
            float4 f = *reinterpret_cast<const float4*>(&x[((size_t)b * HW + (n0 + n)) * 256 + (k0 + k)]);
            *reinterpret_cast<v4bf*>(&Bst[n][k]) = pack4(f.x, f.y, f.z, f.w);
        }
        if (k0 + 32 < 256) {                     // prefetch next K tile
            int g = tid;
            __builtin_prefetch(&w1[(size_t)(m0 + (g >> 3)) * 256 + (k0 + 32) + (g & 7) * 4], 0, 1);
            __builtin_prefetch(&x[((size_t)b * HW + (n0 + (g & 63))) * 256 + (k0 + 32)], 0, 1);
        }
        __syncthreads();

        const v8bf* ar0 = reinterpret_cast<const v8bf*>(&As[wm + lm][0]);
        const v8bf* ar1 = reinterpret_cast<const v8bf*>(&As[wm + 16 + lm][0]);
        const v8bf* br0 = reinterpret_cast<const v8bf*>(&Bst[wn + lm][0]);
        const v8bf* br1 = reinterpret_cast<const v8bf*>(&Bst[wn + 16 + lm][0]);
        v16bf a0  = cat8(ar0[lh], ar0[lh + 2]);       // k = lh*8..+7, 16+lh*8..+7
        v16bf a1  = cat8(ar1[lh], ar1[lh + 2]);
        v16bf b0v = cat8(br0[lh], br0[lh + 2]);  // k = lh*8..+7, 16+lh*8..+7
        v16bf b1v = cat8(br1[lh], br1[lh + 2]);

        acc00 = __builtin_amdgcn_wmma_f32_16x16x32_bf16(false, a0, false, b0v, (short)0, acc00, false, false);
        acc01 = __builtin_amdgcn_wmma_f32_16x16x32_bf16(false, a0, false, b1v, (short)0, acc01, false, false);
        acc10 = __builtin_amdgcn_wmma_f32_16x16x32_bf16(false, a1, false, b0v, (short)0, acc10, false, false);
        acc11 = __builtin_amdgcn_wmma_f32_16x16x32_bf16(false, a1, false, b1v, (short)0, acc11, false, false);
        __syncthreads();
    }

    float* yout = ws + Y_OFF;
    #pragma unroll
    for (int mt = 0; mt < 2; ++mt) {
        #pragma unroll
        for (int r = 0; r < 8; ++r) {
            int m = m0 + wm + mt * 16 + lh * 8 + r;     // output channel
            float bias = b1[m];
            float v0 = (mt == 0) ? acc00[r] : acc10[r];
            float v1 = (mt == 0) ? acc01[r] : acc11[r];
            int na = n0 + wn + lm, nb_ = na + 16;
            float g0 = gelu_exact(v0 + bias);
            float g1 = gelu_exact(v1 + bias);
            yout[((size_t)b * Cc + m) * HW + na]  = g0;
            yout[((size_t)b * Cc + m) * HW + nb_] = g1;
            float s = g0 + g1, ss = g0 * g0 + g1 * g1;
            #pragma unroll
            for (int off = 1; off < 16; off <<= 1) {
                s  += __shfl_xor(s,  off, 16);
                ss += __shfl_xor(ss, off, 16);
            }
            if (lm == 0) {
                atomicAdd(&ws[SUM1 + m], s);
                atomicAdd(&ws[SSQ1 + m], ss);
                atomicAdd(&ws[GAPRAW + (size_t)b * Cc + m], s);
            }
        }
    }
}

// ---------------------------------------------------------------------------
// BN1 affine + GAP -> relu FC -> softmax dynamic kernel weights (tiny)
// ---------------------------------------------------------------------------
__global__ __launch_bounds__(256)
void k_bn1_dyn(const float* __restrict__ g1, const float* __restrict__ be1,
               const float* __restrict__ aw1, const float* __restrict__ ab1,
               const float* __restrict__ aw2, const float* __restrict__ ab2,
               float* __restrict__ ws) {
    int tid = threadIdx.x;
    for (int c = tid; c < Cc; c += 256) {
        float s = ws[SUM1 + c], q = ws[SSQ1 + c];
        float m = s / RED_N, var = q / RED_N - m * m;
        float sc = g1[c] * rsqrtf(var + EPS);
        ws[SCALE1 + c] = sc;
        ws[SHIFT1 + c] = be1[c] - m * sc;
    }
    __syncthreads();
    for (int idx = tid; idx < Bz * 128; idx += 256) {   // h1 = relu(aw1 @ gap + ab1)
        int b = idx >> 7, o = idx & 127;
        float acc = ab1[o];
        for (int c = 0; c < Cc; ++c) {
            float gap = ws[GAPRAW + (size_t)b * Cc + c] * (1.0f / 1024.0f) * ws[SCALE1 + c] + ws[SHIFT1 + c];
            acc += aw1[(size_t)o * Cc + c] * gap;
        }
        ws[H1BUF + idx] = fmaxf(acc, 0.0f);
    }
    __syncthreads();
    __shared__ float logits[8][9];
    for (int idx = tid; idx < 72; idx += 256) {
        int b = idx / 9, j = idx % 9;
        float acc = ab2[j];
        for (int o = 0; o < 128; ++o) acc += aw2[j * 128 + o] * ws[H1BUF + b * 128 + o];
        logits[b][j] = acc;
    }
    __syncthreads();
    if (tid < 8) {                                       // softmax over 9
        float mx = -3.4e38f;
        for (int j = 0; j < 9; ++j) mx = fmaxf(mx, logits[tid][j]);
        float e[9], sum = 0.0f;
        for (int j = 0; j < 9; ++j) { e[j] = expf(logits[tid][j] - mx); sum += e[j]; }
        for (int j = 0; j < 9; ++j) ws[KWBUF + tid * 16 + j] = e[j] / sum;
    }
}

// ---------------------------------------------------------------------------
// Dynamic 3x3 depthwise (BN1 applied on the fly, zero-pad) + GELU.
// One block per (b,c) plane; 34x34 LDS halo; BN2 stats + per-(b,c) sum/max/min
// ---------------------------------------------------------------------------
__global__ __launch_bounds__(256)
void k_dyndw(float* __restrict__ ws) {
    __shared__ float tile[34][35];
    __shared__ float red[256];
    int tid = threadIdx.x;
    int bc = blockIdx.x;                 // 0..8191
    int b = bc >> 10, c = bc & 1023;
    const float* y = ws + Y_OFF + ((size_t)b * Cc + c) * HW;
    float sc = ws[SCALE1 + c], sh = ws[SHIFT1 + c];
    for (int idx = tid; idx < 34 * 34; idx += 256) {
        int py = idx / 34, px = idx % 34;
        int hy = py - 1, wx = px - 1;
        float v = 0.0f;
        if (hy >= 0 && hy < 32 && wx >= 0 && wx < 32) v = y[hy * 32 + wx] * sc + sh;
        tile[py][px] = v;
    }
    float kwv[9];
    #pragma unroll
    for (int j = 0; j < 9; ++j) kwv[j] = ws[KWBUF + b * 16 + j];
    __syncthreads();

    float lsum = 0.0f, lss = 0.0f, lmax = -3.4e38f, lmin = 3.4e38f;
    float* t = ws + T_OFF + ((size_t)b * Cc + c) * HW;
    #pragma unroll
    for (int i = 0; i < 4; ++i) {
        int s = tid + i * 256;
        int oy = s >> 5, ox = s & 31;
        float a = 0.0f;
        #pragma unroll
        for (int ky = 0; ky < 3; ++ky)
            #pragma unroll
            for (int kx = 0; kx < 3; ++kx)
                a += kwv[ky * 3 + kx] * tile[oy + ky][ox + kx];
        float g = gelu_exact(a);
        t[s] = g;
        lsum += g; lss += g * g;
        lmax = fmaxf(lmax, g); lmin = fminf(lmin, g);
    }
    // block reductions (sum, sumsq, max, min)
    red[tid] = lsum; __syncthreads();
    for (int st = 128; st > 0; st >>= 1) { if (tid < st) red[tid] += red[tid + st]; __syncthreads(); }
    float bsum = red[0]; __syncthreads();
    red[tid] = lss; __syncthreads();
    for (int st = 128; st > 0; st >>= 1) { if (tid < st) red[tid] += red[tid + st]; __syncthreads(); }
    float bss = red[0]; __syncthreads();
    red[tid] = lmax; __syncthreads();
    for (int st = 128; st > 0; st >>= 1) { if (tid < st) red[tid] = fmaxf(red[tid], red[tid + st]); __syncthreads(); }
    float bmax = red[0]; __syncthreads();
    red[tid] = lmin; __syncthreads();
    for (int st = 128; st > 0; st >>= 1) { if (tid < st) red[tid] = fminf(red[tid], red[tid + st]); __syncthreads(); }
    if (tid == 0) {
        atomicAdd(&ws[SUM2 + c], bsum);
        atomicAdd(&ws[SSQ2 + c], bss);
        ws[GAP2 + (size_t)b * Cc + c] = bsum;
        ws[MAX2 + (size_t)b * Cc + c] = bmax;
        ws[MIN2 + (size_t)b * Cc + c] = red[0];   // min
    }
}

// ---------------------------------------------------------------------------
// BN2 + channel attention -> fold into per-(b,c) m2/a2; precompute cA[b,d]
// ---------------------------------------------------------------------------
__global__ __launch_bounds__(256)
void k_bn2_catt(const float* __restrict__ g2, const float* __restrict__ be2,
                const float* __restrict__ caw1, const float* __restrict__ caw2,
                const float* __restrict__ pw, float* __restrict__ ws) {
    int tid = threadIdx.x;
    for (int c = tid; c < Cc; c += 256) {
        float s = ws[SUM2 + c], q = ws[SSQ2 + c];
        float m = s / RED_N, var = q / RED_N - m * m;
        float sc = g2[c] * rsqrtf(var + EPS);
        ws[SCALE2 + c] = sc;
        ws[SHIFT2 + c] = be2[c] - m * sc;
    }
    __syncthreads();
    // avg2 -> M2O (temp), mx2 -> A2O (temp); max of affine = max/min raw by sign
    for (int idx = tid; idx < Bz * Cc; idx += 256) {
        int c = idx & 1023;
        float sc = ws[SCALE2 + c], sh = ws[SHIFT2 + c];
        float avg = ws[GAP2 + idx] * (1.0f / 1024.0f) * sc + sh;
        float mx  = (sc >= 0.0f) ? (ws[MAX2 + idx] * sc + sh) : (ws[MIN2 + idx] * sc + sh);
        ws[M2O + idx] = avg;
        ws[A2O + idx] = mx;
    }
    __syncthreads();
    __shared__ float r1s[8][64];     // relu(w1@avg)+relu(w1@mx)
    for (int idx = tid; idx < Bz * 64; idx += 256) {
        int b = idx >> 6, j = idx & 63;
        float aA = 0.0f, aM = 0.0f;
        for (int c = 0; c < Cc; ++c) {
            float w = caw1[(size_t)j * Cc + c];
            aA += w * ws[M2O + (size_t)b * Cc + c];
            aM += w * ws[A2O + (size_t)b * Cc + c];
        }
        r1s[b][j] = fmaxf(aA, 0.0f) + fmaxf(aM, 0.0f);
    }
    __syncthreads();
    for (int idx = tid; idx < Bz * Cc; idx += 256) {
        int b = idx >> 10, c = idx & 1023;
        float acc = 0.0f;
        #pragma unroll 4
        for (int j = 0; j < 64; ++j) acc += caw2[(size_t)c * 64 + j] * r1s[b][j];
        float sgm = 1.0f / (1.0f + expf(-acc));
        ws[M2O + idx] = sgm * ws[SCALE2 + c];
        ws[A2O + idx] = sgm * ws[SHIFT2 + c];
    }
    __syncthreads();
    for (int idx = tid; idx < Bz * Dd; idx += 256) {     // cA[b,d] = sum_c pw*a2
        int b = idx >> 8, dd = idx & 255;
        float acc = 0.0f;
        for (int c = 0; c < Cc; ++c) acc += pw[(size_t)dd * Cc + c] * ws[A2O + (size_t)b * Cc + c];
        ws[CAB + idx] = acc;
    }
}

// ---------------------------------------------------------------------------
// Per-pixel channel mean/max of u = t*m2 + a2
// ---------------------------------------------------------------------------
__global__ __launch_bounds__(256)
void k_spmaps(float* __restrict__ ws) {
    int tid = threadIdx.x;
    int b  = blockIdx.x >> 4;
    int s0 = (blockIdx.x & 15) * 64;
    int crow = tid >> 6, scol = tid & 63;
    const float* t = ws + T_OFF + (size_t)b * Cc * HW;
    float acc = 0.0f, amax = -3.4e38f;
    for (int c = crow; c < Cc; c += 4) {
        float u = t[(size_t)c * HW + s0 + scol] * ws[M2O + (size_t)b * Cc + c] + ws[A2O + (size_t)b * Cc + c];
        acc += u; amax = fmaxf(amax, u);
    }
    __shared__ float ssum[4][64], smax[4][64];
    ssum[crow][scol] = acc; smax[crow][scol] = amax;
    __syncthreads();
    if (tid < 64) {
        float st = ssum[0][tid] + ssum[1][tid] + ssum[2][tid] + ssum[3][tid];
        float mt = fmaxf(fmaxf(smax[0][tid], smax[1][tid]), fmaxf(smax[2][tid], smax[3][tid]));
        ws[AVGM + (size_t)b * HW + s0 + tid] = st * (1.0f / 1024.0f);
        ws[MAXM + (size_t)b * HW + s0 + tid] = mt;
    }
}

// 7x7 spatial conv (zero pad 3) + sigmoid
__global__ __launch_bounds__(256)
void k_spconv(const float* __restrict__ sw, const float* __restrict__ sb, float* __restrict__ ws) {
    int idx = blockIdx.x * 256 + threadIdx.x;     // 8192
    int b = idx >> 10, s = idx & 1023;
    int oy = s >> 5, ox = s & 31;
    float acc = sb[0];
    for (int i = 0; i < 7; ++i) {
        int yy = oy + i - 3;
        if (yy < 0 || yy >= 32) continue;
        for (int j = 0; j < 7; ++j) {
            int xx = ox + j - 3;
            if (xx < 0 || xx >= 32) continue;
            int p = b * HW + yy * 32 + xx;
            acc += sw[i * 7 + j] * ws[AVGM + p] + sw[49 + i * 7 + j] * ws[MAXM + p];
        }
    }
    ws[SPM + idx] = 1.0f / (1.0f + expf(-acc));
}

// ---------------------------------------------------------------------------
// GEMM2: v[b,d,s] = sp[b,s]*( sum_c (pw*m2) t[b,c,s] + cA[b,d] ) + pb[d]
// Same tiling as GEMM1; K=1024. Epilogue accumulates BN3 stats.
// ---------------------------------------------------------------------------
__global__ __launch_bounds__(256)
void k_gemm2(const float* __restrict__ pw, const float* __restrict__ pb, float* __restrict__ ws) {
    __shared__ __align__(64) __bf16 As[128][32];
    __shared__ __align__(64) __bf16 Bst[64][32];
    const int tid = threadIdx.x;
    const int b   = blockIdx.z;
    const int m0  = blockIdx.x * 128;
    const int n0  = blockIdx.y * 64;
    const int wave = tid >> 5, lane = tid & 31;
    const int lm = lane & 15, lh = lane >> 4;
    const int wm = (wave & 3) * 32, wn = (wave >> 2) * 32;
    const float* t  = ws + T_OFF + (size_t)b * Cc * HW;
    const float* m2 = ws + M2O + (size_t)b * Cc;

    v8f acc00 = {}, acc01 = {}, acc10 = {}, acc11 = {};

    for (int k0 = 0; k0 < Cc; k0 += 32) {
        // A: pw * m2 folded at load, float4 along k
        #pragma unroll
        for (int i = 0; i < 4; ++i) {
            int g = tid + i * 256;
            int m = g >> 3, k = (g & 7) * 4;
            float4 f = *reinterpret_cast<const float4*>(&pw[(size_t)(m0 + m) * Cc + (k0 + k)]);
            float4 q = *reinterpret_cast<const float4*>(&m2[k0 + k]);
            *reinterpret_cast<v4bf*>(&As[m][k]) = pack4(f.x * q.x, f.y * q.y, f.z * q.z, f.w * q.w);
        }
        // B: t[k][n] contiguous along n; float4 along n, scatter to Bst[n][k]
        #pragma unroll
        for (int i = 0; i < 2; ++i) {
            int g = tid + i * 256;               // 0..511
            int k = g >> 4, n = (g & 15) * 4;
            float4 f = *reinterpret_cast<const float4*>(&t[(size_t)(k0 + k) * HW + (n0 + n)]);
            Bst[n + 0][k] = (__bf16)f.x;
            Bst[n + 1][k] = (__bf16)f.y;
            Bst[n + 2][k] = (__bf16)f.z;
            Bst[n + 3][k] = (__bf16)f.w;
        }
        if (k0 + 32 < Cc) {                      // prefetch next K tile
            int g = tid;
            __builtin_prefetch(&pw[(size_t)(m0 + (g >> 3)) * Cc + (k0 + 32) + (g & 7) * 4], 0, 1);
            __builtin_prefetch(&t[(size_t)(k0 + 32 + (g >> 4)) * HW + n0 + (g & 15) * 4], 0, 1);
        }
        __syncthreads();

        const v8bf* ar0 = reinterpret_cast<const v8bf*>(&As[wm + lm][0]);
        const v8bf* ar1 = reinterpret_cast<const v8bf*>(&As[wm + 16 + lm][0]);
        const v8bf* br0 = reinterpret_cast<const v8bf*>(&Bst[wn + lm][0]);
        const v8bf* br1 = reinterpret_cast<const v8bf*>(&Bst[wn + 16 + lm][0]);
        v16bf a0  = cat8(ar0[lh], ar0[lh + 2]);
        v16bf a1  = cat8(ar1[lh], ar1[lh + 2]);
        v16bf b0v = cat8(br0[lh], br0[lh + 2]);
        v16bf b1v = cat8(br1[lh], br1[lh + 2]);

        acc00 = __builtin_amdgcn_wmma_f32_16x16x32_bf16(false, a0, false, b0v, (short)0, acc00, false, false);
        acc01 = __builtin_amdgcn_wmma_f32_16x16x32_bf16(false, a0, false, b1v, (short)0, acc01, false, false);
        acc10 = __builtin_amdgcn_wmma_f32_16x16x32_bf16(false, a1, false, b0v, (short)0, acc10, false, false);
        acc11 = __builtin_amdgcn_wmma_f32_16x16x32_bf16(false, a1, false, b1v, (short)0, acc11, false, false);
        __syncthreads();
    }

    #pragma unroll
    for (int mt = 0; mt < 2; ++mt) {
        #pragma unroll
        for (int r = 0; r < 8; ++r) {
            int m = m0 + wm + mt * 16 + lh * 8 + r;         // output dim d
            float pbm = pb[m];
            float ca  = ws[CAB + (size_t)b * Dd + m];
            int na = n0 + wn + lm, nb_ = na + 16;
            float sp0 = ws[SPM + (size_t)b * HW + na];
            float sp1 = ws[SPM + (size_t)b * HW + nb_];
            float v0 = sp0 * (((mt == 0) ? acc00[r] : acc10[r]) + ca) + pbm;
            float v1 = sp1 * (((mt == 0) ? acc01[r] : acc11[r]) + ca) + pbm;
            ws[V_OFF + ((size_t)b * Dd + m) * HW + na]  = v0;
            ws[V_OFF + ((size_t)b * Dd + m) * HW + nb_] = v1;
            float s = v0 + v1, ss = v0 * v0 + v1 * v1;
            #pragma unroll
            for (int off = 1; off < 16; off <<= 1) {
                s  += __shfl_xor(s,  off, 16);
                ss += __shfl_xor(ss, off, 16);
            }
            if (lm == 0) {
                atomicAdd(&ws[SUM3 + m], s);
                atomicAdd(&ws[SSQ3 + m], ss);
            }
        }
    }
}

__global__ void k_bn3prep(const float* __restrict__ g3, const float* __restrict__ be3,
                          float* __restrict__ ws) {
    int dd = threadIdx.x;        // 256 threads
    float s = ws[SUM3 + dd], q = ws[SSQ3 + dd];
    float m = s / RED_N, var = q / RED_N - m * m;
    float sc = g3[dd] * rsqrtf(var + EPS);
    ws[SCALE3 + dd] = sc;
    ws[SHIFT3 + dd] = be3[dd] - m * sc;
}

// out[b,s,d] = x[b,s,d] + bn3(v)[b,d,s]
__global__ __launch_bounds__(256)
void k_final(const float* __restrict__ x, const float* __restrict__ ws, float* __restrict__ out) {
    size_t idx = (size_t)blockIdx.x * 256 + threadIdx.x;   // 2,097,152
    int dd = (int)(idx & 255);
    size_t sb_ = idx >> 8;
    int s = (int)(sb_ & 1023), b = (int)(sb_ >> 10);
    float v = ws[V_OFF + ((size_t)b * Dd + dd) * HW + s];
    out[idx] = x[idx] + v * ws[SCALE3 + dd] + ws[SHIFT3 + dd];
}

// ---------------------------------------------------------------------------
extern "C" void kernel_launch(void* const* d_in, const int* in_sizes, int n_in,
                              void* d_out, int out_size, void* d_ws, size_t ws_size,
                              hipStream_t stream) {
    (void)in_sizes; (void)n_in; (void)out_size; (void)ws_size;
    const float* x    = (const float*)d_in[0];
    const float* w1   = (const float*)d_in[1];
    const float* b1   = (const float*)d_in[2];
    const float* g1   = (const float*)d_in[3];
    const float* be1  = (const float*)d_in[4];
    const float* aw1  = (const float*)d_in[5];
    const float* ab1  = (const float*)d_in[6];
    const float* aw2  = (const float*)d_in[7];
    const float* ab2  = (const float*)d_in[8];
    const float* g2   = (const float*)d_in[9];
    const float* be2  = (const float*)d_in[10];
    const float* caw1 = (const float*)d_in[11];
    const float* caw2 = (const float*)d_in[12];
    const float* pw   = (const float*)d_in[13];
    const float* pb   = (const float*)d_in[14];
    const float* g3   = (const float*)d_in[15];
    const float* be3  = (const float*)d_in[16];
    const float* sw   = (const float*)d_in[17];
    const float* sb   = (const float*)d_in[18];
    float* ws  = (float*)d_ws;
    float* out = (float*)d_out;

    {   // zero atomic accumulators / stats region (graph-replay safe)
        int n = (int)(WS_END - SUM1);
        k_zero<<<(n + 255) / 256, 256, 0, stream>>>(ws + SUM1, n);
    }
    k_gemm1  <<<dim3(Cc / 128, HW / 64, Bz), 256, 0, stream>>>(x, w1, b1, ws);
    k_bn1_dyn<<<1, 256, 0, stream>>>(g1, be1, aw1, ab1, aw2, ab2, ws);
    k_dyndw  <<<Bz * Cc, 256, 0, stream>>>(ws);
    k_bn2_catt<<<1, 256, 0, stream>>>(g2, be2, caw1, caw2, pw, ws);
    k_spmaps <<<Bz * (HW / 64), 256, 0, stream>>>(ws);
    k_spconv <<<(Bz * HW) / 256, 256, 0, stream>>>(sw, sb, ws);
    k_gemm2  <<<dim3(Dd / 128, HW / 64, Bz), 256, 0, stream>>>(pw, pb, ws);
    k_bn3prep<<<1, 256, 0, stream>>>(g3, be3, ws);
    k_final  <<<(Bz * HW * Dd) / 256, 256, 0, stream>>>(x, ws, out);
}
